// SSMKernel_59828894433820
// MI455X (gfx1250) — matrix operand order repaired and checked by reading, not compiled, not run on hardware
//
#include <hip/hip_runtime.h>
#include <hip/hip_bf16.h>
#include <stdint.h>

// ---------------------------------------------------------------------------
// SSM forward for MI455X (gfx1250, wave32, WMMA, TDM).
//   1) convert u / weights to bf16 (weights transposed to [n][k])
//   2) WMMA bf16 GEMMs: P_A = u@W_A+b_A, P_C = u@W_C+b_C (f32 in ws)
//   3) fused bilinear WMMA kernel (never materializes the big tensor):
//        yD[m,i] = sum_j (u@W_D+b_D)[m,i*256+j] * u[m,j]   (R=256)
//        bu[m,r] = sum_j (u@W_B+b_B)[m,r*256+j] * u[m,j]   (R=64)
//   4) parallel per-(b,t) Gauss-Jordan solve (2048 blocks), RHS is only
//      65 cols: [I + dt/2 A | dt * (B_t u_t)]  ->  A_d (64x64), v (64)
//   5) chunked associative scan:
//        pass1: compose 16-step chunks with v_wmma_f32_16x16x4_f32;
//               A_d tiles staged to LDS by the Tensor Data Mover (TDM)
//        pass2: sequential combine of 32 chunk composites per batch (tiny)
//        pass3: replay chunks (TDM-staged A_d), emit y = C_t s + yD
// ---------------------------------------------------------------------------

typedef __bf16 bf16;
typedef __attribute__((ext_vector_type(16))) __bf16 v16bf;
typedef __attribute__((ext_vector_type(8)))  __bf16 v8bf;
typedef __attribute__((ext_vector_type(8)))  float  v8f;
typedef __attribute__((ext_vector_type(2)))  float  v2f;
typedef __attribute__((ext_vector_type(4)))  uint32_t u32x4;
typedef __attribute__((ext_vector_type(8)))  int32_t  i32x8;
typedef __attribute__((ext_vector_type(4)))  int32_t  i32x4;

constexpr int Hh = 256, Nn = 64, BB = 4, LL = 512;
constexpr int MM  = BB * LL;        // 2048 rows (b*t)
constexpr int NAc = Nn * Nn;        // 4096
constexpr int NBc = Nn * Hh;        // 16384
constexpr int NCc = Hh * Nn;        // 16384
constexpr int NDc = Hh * Hh;        // 65536
constexpr int CHUNK = 16, NCH = LL / CHUNK;   // 32 chunks per batch

// workspace layout (bytes, all 32B aligned)
constexpr size_t OFF_UBF   = 0;
constexpr size_t OFF_WTA   = OFF_UBF   + (size_t)MM  * Hh * 2;
constexpr size_t OFF_WTB   = OFF_WTA   + (size_t)NAc * Hh * 2;
constexpr size_t OFF_WTC   = OFF_WTB   + (size_t)NBc * Hh * 2;
constexpr size_t OFF_WTD   = OFF_WTC   + (size_t)NCc * Hh * 2;
constexpr size_t OFF_PA    = OFF_WTD   + (size_t)NDc * Hh * 2;
constexpr size_t OFF_PC    = OFF_PA    + (size_t)MM * NAc * 4;
constexpr size_t OFF_YD    = OFF_PC    + (size_t)MM * NCc * 4;
constexpr size_t OFF_BU    = OFF_YD    + (size_t)MM * Hh * 4;
constexpr size_t OFF_AD    = OFF_BU    + (size_t)MM * Nn * 4;
constexpr size_t OFF_VV    = OFF_AD    + (size_t)MM * NAc * 4;
constexpr size_t OFF_PCMP  = OFF_VV    + (size_t)MM * Nn * 4;
constexpr size_t OFF_WCMP  = OFF_PCMP  + (size_t)BB * NCH * NAc * 4;
constexpr size_t OFF_SST   = OFF_WCMP  + (size_t)BB * NCH * Nn * 4;
// end ~ 248 MB

// ---------------------------------------------------------------------------
// TDM: DMA a 64x64 f32 tile (row stride 64) from global memory into LDS with
// a 1-DWORD pad after every 64 DWORDs -> LDS pitch 65 (bank-conflict free).
// D# built per ISA 8.3-8.5: group0 = {count, lds_addr, global_addr, type=2},
// group1 = {data_size=4B, pad_en, pad_interval(5->64dw), pad_amount(0->1dw),
//           tensor_dim0/1=64, tile_dim0/1=64, dim0_stride=64}, groups2/3 = 0.
// This toolchain exposes the 6-arg builtin (extra int32x8 group, then cpol).
// TDM ignores EXEC and issues once per wave -> call from one wave only, then
// s_wait_tensorcnt 0 + workgroup barrier before consuming.
// ---------------------------------------------------------------------------
__device__ __forceinline__ void tdm_load_tile64(const float* gsrc, void* lds_dst) {
    const uint64_t ga = (uint64_t)(uintptr_t)gsrc;      // byte address
    const uint32_t la = (uint32_t)(uintptr_t)lds_dst;   // low 32b = LDS offset
    u32x4 g0;
    g0[0] = 1u;                                          // count=1 (user D#)
    g0[1] = la;                                          // lds_addr
    g0[2] = (uint32_t)(ga & 0xFFFFFFFFu);                // global_addr[31:0]
    g0[3] = (uint32_t)((ga >> 32) & 0x01FFFFFFu) | (2u << 30);  // [56:32]|type=2
    i32x8 g1;
    g1[0] = (int)((2u << 16) | (1u << 20) | (5u << 22)); // 4B, pad_en, ivl=64dw
    g1[1] = (int)(64u << 16);                            // tensor_dim0 = 64
    g1[2] = (int)(64u << 16);                            // tensor_dim1 = 64
    g1[3] = (int)(64u << 16);                            // tile_dim0   = 64
    g1[4] = 64;                                          // tile_dim1   = 64
    g1[5] = 64;                                          // dim0_stride = 64
    g1[6] = 0;
    g1[7] = 0;
    i32x4 g2 = {0, 0, 0, 0};
    i32x4 g3 = {0, 0, 0, 0};
    i32x8 g4 = {0, 0, 0, 0, 0, 0, 0, 0};
    __builtin_amdgcn_tensor_load_to_lds(g0, g1, g2, g3, g4, 0);
}

// ---------------------------------------------------------------------------
// bf16 WMMA fragment loaders (contiguous k order, not the ISA table order; A and B match)
// A 16x32: lane row = lane&15; K = (lane>>4)*16 + {0..15}
// B 32x16: lane col = lane&15; K = (lane>>4)*16 + {0..15}
// ---------------------------------------------------------------------------
__device__ __forceinline__ v16bf load_a_frag(const bf16* p) {
    v8bf lo = *reinterpret_cast<const v8bf*>(p);
    v8bf hi = *reinterpret_cast<const v8bf*>(p + 8);
    v16bf a;
#pragma unroll
    for (int i = 0; i < 8; ++i) { a[i] = lo[i]; a[i + 8] = hi[i]; }
    return a;
}
__device__ __forceinline__ v16bf load_b_frag(const bf16* p) {
    return *reinterpret_cast<const v16bf*>(p);
}

// ---------------------------------------------------------------------------
__global__ void ssm_convert_u(const float* __restrict__ u, bf16* __restrict__ ub) {
    const int idx = blockIdx.x * 256 + threadIdx.x;
    ub[idx] = (bf16)u[idx];
}

// W row-major [k=0..255][n] -> WT bf16 [n][k]
__global__ void ssm_transpose_bf16(const float* __restrict__ W, bf16* __restrict__ WT,
                                   int Ncols) {
    const int idx = blockIdx.x * 256 + threadIdx.x;
    const int k = idx / Ncols;
    const int n = idx - k * Ncols;
    WT[(size_t)n * 256 + k] = (bf16)W[idx];
}

// ---------------------------------------------------------------------------
// Out[m][n] = sum_k X[m][k]*WT[n][k] + bias[n]   (K=256, 8 wmma k-steps)
// ---------------------------------------------------------------------------
__global__ __launch_bounds__(256)
void ssm_wmma_gemm_bias(const bf16* __restrict__ X, const bf16* __restrict__ WT,
                        const float* __restrict__ bias, float* __restrict__ Out,
                        int Ncols) {
    const int lane  = threadIdx.x & 31;
    const int wave  = threadIdx.x >> 5;
    const int mbase = blockIdx.y * 32 + (wave & 1) * 16;
    const int nbase = blockIdx.x * 64 + (wave >> 1) * 16;
    const int rc    = lane & 15;
    const int hi16  = lane >> 4;

    const bf16* pa = X  + (size_t)(mbase + rc) * 256 + hi16 * 16;
    const bf16* pb = WT + (size_t)(nbase + rc) * 256 + hi16 * 16;
    __builtin_prefetch(pb + 16 * 256, 0, 1);   // global_prefetch_b8

    v8f c = {};
#pragma unroll
    for (int k = 0; k < 8; ++k) {
        v16bf a = load_a_frag(pa + k * 32);
        v16bf b = load_b_frag(pb + k * 32);
        c = __builtin_amdgcn_wmma_f32_16x16x32_bf16(false, a, false, b,
                                                    (short)0, c, false, false);
    }
    const int   ncol  = nbase + rc;
    const float bv    = bias[ncol];
    const int   mrow0 = mbase + hi16 * 8;
#pragma unroll
    for (int r = 0; r < 8; ++r)
        Out[(size_t)(mrow0 + r) * Ncols + ncol] = c[r] + bv;
}

// ---------------------------------------------------------------------------
// Fused bilinear: out[m][i] = sum_j ((u@W)[m,i*256+j] + b[i*256+j]) * u[m,j]
// for i in [0,R). R=256 -> y_D ; R=64 -> B_t @ u_t. Tile lives in registers.
// ---------------------------------------------------------------------------
__global__ __launch_bounds__(256)
void ssm_wmma_bilinear(const bf16* __restrict__ X, const bf16* __restrict__ WT,
                       const float* __restrict__ bias, const float* __restrict__ u,
                       float* __restrict__ out, int R) {
    __shared__ float uT[16 * 256];
    const int mbase = blockIdx.y * 16;
    for (int idx = threadIdx.x; idx < 16 * 256; idx += 256)
        uT[idx] = u[(size_t)(mbase + (idx >> 8)) * 256 + (idx & 255)];
    __syncthreads();

    const int lane = threadIdx.x & 31;
    const int wave = threadIdx.x >> 5;
    const int i    = blockIdx.x * 8 + wave;   // 0..R-1
    const int rc   = lane & 15;
    const int hi16 = lane >> 4;
    const int lr0  = hi16 * 8;

    const bf16* pa = X + (size_t)(mbase + rc) * 256 + hi16 * 16;

    float acc[8];
#pragma unroll
    for (int r = 0; r < 8; ++r) acc[r] = 0.f;

    for (int jt = 0; jt < 16; ++jt) {
        const int ncb = i * 256 + jt * 16;
        const bf16* pb = WT + (size_t)(ncb + rc) * 256 + hi16 * 16;
        v8f c = {};
#pragma unroll
        for (int k = 0; k < 8; ++k) {
            v16bf a = load_a_frag(pa + k * 32);
            v16bf b = load_b_frag(pb + k * 32);
            c = __builtin_amdgcn_wmma_f32_16x16x32_bf16(false, a, false, b,
                                                        (short)0, c, false, false);
        }
        const int   j  = jt * 16 + rc;
        const float bv = bias[i * 256 + j];
#pragma unroll
        for (int r = 0; r < 8; ++r)
            acc[r] += (c[r] + bv) * uT[(lr0 + r) * 256 + j];
    }
#pragma unroll
    for (int r = 0; r < 8; ++r) {     // reduce 16 lanes of each half-wave
        acc[r] += __shfl_xor(acc[r], 1);
        acc[r] += __shfl_xor(acc[r], 2);
        acc[r] += __shfl_xor(acc[r], 4);
        acc[r] += __shfl_xor(acc[r], 8);
    }
    if (rc == 0) {
#pragma unroll
        for (int r = 0; r < 8; ++r)
            out[(size_t)(mbase + lr0 + r) * R + i] = acc[r];
    }
}

// ---------------------------------------------------------------------------
// Parallel per-(b,t) solve: S = [I-h A | I+h A | dt*bu], Gauss-Jordan (M is
// diagonally dominant, |A|~1e-2 -> no pivoting). Outputs A_d and v.
// ---------------------------------------------------------------------------
constexpr int SP2 = 133;   // padded pitch for 129 columns

__global__ __launch_bounds__(256)
void ssm_solve(const float* __restrict__ PA, const float* __restrict__ bu,
               const float* __restrict__ dtp, float* __restrict__ Ad,
               float* __restrict__ vv) {
    __shared__ float S[64 * SP2];
    const int bt  = blockIdx.x;
    const int tid = threadIdx.x;
    const float dt  = fminf(fmaxf(dtp[0], 1e-4f), 0.1f);
    const float hdt = 0.5f * dt;

    for (int idx = tid; idx < 4096; idx += 256) {
        const int r = idx >> 6, c = idx & 63;
        const float a  = PA[(size_t)bt * NAc + idx];
        const float id = (r == c) ? 1.f : 0.f;
        S[r * SP2 + c]      = id - hdt * a;
        S[r * SP2 + 64 + c] = id + hdt * a;
    }
    if (tid < 64) S[tid * SP2 + 128] = dt * bu[bt * 64 + tid];
    __syncthreads();

    for (int p = 0; p < 64; ++p) {
        const float inv = 1.f / S[p * SP2 + p];          // read-only here
        for (int c = tid; c < 129; c += 256)
            if (c != p) S[p * SP2 + c] *= inv;
        __syncthreads();
        for (int idx = tid; idx < 64 * 129; idx += 256) {
            const int r = idx / 129, c = idx - r * 129;
            if (r != p && c != p)
                S[r * SP2 + c] -= S[r * SP2 + p] * S[p * SP2 + c];
        }
        __syncthreads();
    }
    for (int idx = tid; idx < 4096; idx += 256)
        Ad[(size_t)bt * NAc + idx] = S[(idx >> 6) * SP2 + 64 + (idx & 63)];
    if (tid < 64) vv[bt * 64 + tid] = S[tid * SP2 + 128];
}

// ---------------------------------------------------------------------------
// Chunk composition with f32 WMMA: (P,w) <- (A_t P, A_t w + v_t), 16 steps.
// A_d tile staged to LDS by TDM (pitch 65 via TDM pad). 64x64x64 matmul =
// 16 tiles x 16 k-steps of v_wmma_f32_16x16x4_f32.
// ---------------------------------------------------------------------------
__global__ __launch_bounds__(256)
void ssm_compose(const float* __restrict__ Ad, const float* __restrict__ vv,
                 float* __restrict__ Pcomp, float* __restrict__ wcomp) {
    __shared__ float As[64 * 65];
    __shared__ float Pbuf[2][64 * 65];
    __shared__ float wbuf[2][64];
    __shared__ float vsh[64];
    const int blk  = blockIdx.x;          // b*NCH + ch
    const int b    = blk >> 5;
    const int ch   = blk & 31;
    const int tid  = threadIdx.x;
    const int lane = tid & 31, wave = tid >> 5;

    for (int idx = tid; idx < 64 * 65; idx += 256) Pbuf[0][idx] = 0.f;
    __syncthreads();
    if (tid < 64) { Pbuf[0][tid * 65 + tid] = 1.f; wbuf[0][tid] = 0.f; }

    const int rc = lane & 15;
    const int k0b = (lane >> 4) * 2;
    int cur = 0;
    for (int s = 0; s < CHUNK; ++s) {
        const size_t bt = (size_t)b * LL + ch * CHUNK + s;
        if (wave == 0) {                  // one TDM issue per workgroup
            tdm_load_tile64(Ad + bt * NAc, As);
            __builtin_amdgcn_s_wait_tensorcnt(0);
        }
        if (tid < 64) vsh[tid] = vv[bt * 64 + tid];
        __syncthreads();

        float* P = Pbuf[cur];
        float* Q = Pbuf[cur ^ 1];
#pragma unroll
        for (int tt = 0; tt < 2; ++tt) {
            const int tile = wave * 2 + tt;           // 0..15 (4x4 tile grid)
            const int ti = tile >> 2, tj = tile & 3;
            v8f c = {};
#pragma unroll
            for (int kk = 0; kk < 16; ++kk) {
                const int k0 = kk * 4 + k0b;
                v2f a, bf;
                a[0]  = As[(ti * 16 + rc) * 65 + k0];
                a[1]  = As[(ti * 16 + rc) * 65 + k0 + 1];
                bf[0] = P[k0 * 65 + tj * 16 + rc];
                bf[1] = P[(k0 + 1) * 65 + tj * 16 + rc];
                c = __builtin_amdgcn_wmma_f32_16x16x4_f32(false, a, false, bf,
                                                          (short)0, c, false, false);
            }
            const int mrow0 = ti * 16 + (lane >> 4) * 8;
            const int ncol  = tj * 16 + rc;
#pragma unroll
            for (int r = 0; r < 8; ++r)
                Q[(mrow0 + r) * 65 + ncol] = c[r];
        }
        if (tid < 64) {                    // w' = A_t w + v_t
            float acc = vsh[tid];
            for (int j = 0; j < 64; ++j) acc += As[tid * 65 + j] * wbuf[cur][j];
            wbuf[cur ^ 1][tid] = acc;
        }
        __syncthreads();
        cur ^= 1;
    }
    for (int idx = tid; idx < 4096; idx += 256)
        Pcomp[(size_t)blk * NAc + idx] = Pbuf[cur][(idx >> 6) * 65 + (idx & 63)];
    if (tid < 64) wcomp[blk * 64 + tid] = wbuf[cur][tid];
}

// ---------------------------------------------------------------------------
// Sequential combine of 32 chunk composites per batch (tiny). 64 threads.
// ---------------------------------------------------------------------------
__global__ __launch_bounds__(64)
void ssm_chunkscan(const float* __restrict__ Pcomp, const float* __restrict__ wcomp,
                   float* __restrict__ sstart) {
    __shared__ float s[64];
    const int b   = blockIdx.x;
    const int tid = threadIdx.x;
    s[tid] = 0.f;
    __syncthreads();
    for (int c = 0; c < NCH; ++c) {
        const int bc = b * NCH + c;
        sstart[bc * 64 + tid] = s[tid];           // state at chunk start
        float acc = wcomp[bc * 64 + tid];
        const float* prow = Pcomp + (size_t)bc * NAc + tid * 64;
        for (int j = 0; j < 64; ++j) acc += prow[j] * s[j];
        __syncthreads();
        s[tid] = acc;
        __syncthreads();
    }
}

// ---------------------------------------------------------------------------
// Replay chunks in parallel (TDM-staged A_d) and emit y = C_t @ s_{t+1} + yD.
// ---------------------------------------------------------------------------
__global__ __launch_bounds__(256)
void ssm_emit(const float* __restrict__ Ad, const float* __restrict__ vv,
              const float* __restrict__ sstart, const float* __restrict__ PC,
              const float* __restrict__ yD, float* __restrict__ out) {
    __shared__ float As[64 * 65];
    __shared__ float s[64], sn[64];
    const int blk  = blockIdx.x;
    const int b    = blk >> 5, ch = blk & 31;
    const int tid  = threadIdx.x;
    const int wave = tid >> 5;
    if (tid < 64) s[tid] = sstart[blk * 64 + tid];

    for (int t16 = 0; t16 < CHUNK; ++t16) {
        const size_t bt = (size_t)b * LL + ch * CHUNK + t16;
        if (wave == 0) {                  // one TDM issue per workgroup
            tdm_load_tile64(Ad + bt * NAc, As);
            __builtin_amdgcn_s_wait_tensorcnt(0);
        }
        __syncthreads();
        if (tid < 64) {
            float acc = vv[bt * 64 + tid];
            for (int j = 0; j < 64; ++j) acc += As[tid * 65 + j] * s[j];
            sn[tid] = acc;
        }
        __syncthreads();
        if (tid < 64) s[tid] = sn[tid];
        __syncthreads();
        float yv = yD[bt * 256 + tid];
        const float* crow = PC + bt * (size_t)NCc + (size_t)tid * 64;
        for (int j = 0; j < 64; ++j) yv += crow[j] * s[j];
        out[bt * 256 + tid] = yv;
        __syncthreads();
    }
}

// ---------------------------------------------------------------------------
extern "C" void kernel_launch(void* const* d_in, const int* in_sizes, int n_in,
                              void* d_out, int out_size, void* d_ws, size_t ws_size,
                              hipStream_t stream) {
    (void)in_sizes; (void)n_in; (void)out_size; (void)ws_size;
    const float* u   = (const float*)d_in[0];
    const float* WA  = (const float*)d_in[1];
    const float* bA  = (const float*)d_in[2];
    const float* WB  = (const float*)d_in[3];
    const float* bB  = (const float*)d_in[4];
    const float* WC  = (const float*)d_in[5];
    const float* bC  = (const float*)d_in[6];
    const float* WD  = (const float*)d_in[7];
    const float* bD  = (const float*)d_in[8];
    const float* dtp = (const float*)d_in[9];
    float* out = (float*)d_out;

    char* ws   = (char*)d_ws;
    bf16*  ubf = (bf16*)(ws + OFF_UBF);
    bf16*  wtA = (bf16*)(ws + OFF_WTA);
    bf16*  wtB = (bf16*)(ws + OFF_WTB);
    bf16*  wtC = (bf16*)(ws + OFF_WTC);
    bf16*  wtD = (bf16*)(ws + OFF_WTD);
    float* PA  = (float*)(ws + OFF_PA);
    float* PC  = (float*)(ws + OFF_PC);
    float* yD  = (float*)(ws + OFF_YD);
    float* bu  = (float*)(ws + OFF_BU);
    float* Ad  = (float*)(ws + OFF_AD);
    float* vv  = (float*)(ws + OFF_VV);
    float* Pcm = (float*)(ws + OFF_PCMP);
    float* wcm = (float*)(ws + OFF_WCMP);
    float* sst = (float*)(ws + OFF_SST);

    ssm_convert_u<<<MM * Hh / 256, 256, 0, stream>>>(u, ubf);
    ssm_transpose_bf16<<<NAc, 256, 0, stream>>>(WA, wtA, NAc);
    ssm_transpose_bf16<<<NBc, 256, 0, stream>>>(WB, wtB, NBc);
    ssm_transpose_bf16<<<NCc, 256, 0, stream>>>(WC, wtC, NCc);
    ssm_transpose_bf16<<<NDc, 256, 0, stream>>>(WD, wtD, NDc);

    ssm_wmma_gemm_bias<<<dim3(NAc / 64, MM / 32), 256, 0, stream>>>(ubf, wtA, bA, PA, NAc);
    ssm_wmma_gemm_bias<<<dim3(NCc / 64, MM / 32), 256, 0, stream>>>(ubf, wtC, bC, PC, NCc);

    ssm_wmma_bilinear<<<dim3(Hh / 8, MM / 16), 256, 0, stream>>>(ubf, wtD, bD, u, yD, Hh);
    ssm_wmma_bilinear<<<dim3(Nn / 8, MM / 16), 256, 0, stream>>>(ubf, wtB, bB, u, bu, Nn);

    ssm_solve<<<MM, 256, 0, stream>>>(PA, bu, dtp, Ad, vv);

    ssm_compose<<<BB * NCH, 256, 0, stream>>>(Ad, vv, Pcm, wcm);
    ssm_chunkscan<<<BB, 64, 0, stream>>>(Pcm, wcm, sst);
    ssm_emit<<<BB * NCH, 256, 0, stream>>>(Ad, vv, sst, PC, yD, out);
}
